// RFNet_8993661518322
// MI455X (gfx1250) — matrix operand order repaired and checked by reading, not compiled, not run on hardware
//
#include <hip/hip_runtime.h>
#include <hip/hip_bf16.h>

// Problem constants (match reference)
#define B_  2048
#define D_  512
#define E_  100
#define M_  256
#define H_  1024
#define O_  128

// Tiling
#define BT   64          // rows of x per workgroup
#define HC   32          // H-chunk width
#define NT   256         // threads per block (8 wave32)

// Padded LDS strides (elements); all strides * 2B are multiples of 16B
#define GS   256         // g tile row stride (64 x 256)
#define W1S  264         // W1 chunk col-major [n][k], k-stride padded
#define W2S  40          // W2 chunk col-major [o][k]
#define HS   40          // h chunk row-major [r][k]

typedef __attribute__((ext_vector_type(16))) __bf16        v16bf;
typedef __attribute__((ext_vector_type(8)))  float         v8f;
typedef __attribute__((ext_vector_type(4)))  unsigned int  v4u;

union FragU { v16bf f; v4u u[2]; };

// A-matrix (16x32 bf16, row-major source, K contiguous) per ISA 7.12.2:
// lane holds row = lane&15; K-base = (lane&16)?8:0; elements j<8 -> K=kb+j,
// j>=8 -> K=kb+8+j  => two contiguous 16B runs at kb and kb+16.
__device__ inline v16bf load_frag_a(const __bf16* base, int stride, int lane) {
    int row = lane & 15;
    int kb  = (lane & 16) ? 8 : 0;
    FragU t;
    t.u[0] = *(const v4u*)(base + row * stride + kb);
    t.u[1] = *(const v4u*)(base + row * stride + kb + 16);
    return t.f;
}

// B-matrix (32x16 bf16) from col-major [n][k] LDS staging: lane holds
// col n = lane&15; K-base = (lane&16)?8:0; elements j<8 -> K=kb+j,
// j>=8 -> K=kb+8+j  => two contiguous 16B runs at kb and kb+16 (same as A).
__device__ inline v16bf load_frag_b(const __bf16* base, int stride, int lane) {
    int n  = lane & 15;
    int kb = (lane & 16) ? 8 : 0;
    FragU t;
    t.u[0] = *(const v4u*)(base + n * stride + kb);
    t.u[1] = *(const v4u*)(base + n * stride + kb + 16);
    return t.f;
}

__global__ void rfnet_zero_out(float* out, int n) {
    int i = blockIdx.x * blockDim.x + threadIdx.x;
    if (i < n) out[i] = 0.0f;
}

// One-time convert+transpose: src f32 (E batches of R x C, row-major) ->
// dst bf16 (E batches of C x R, row-major). Tiled through LDS so both the
// f32 reads and the bf16 writes are coalesced. R, C multiples of 32.
__global__ __launch_bounds__(256)
void rfnet_transpose_cvt(const float* __restrict__ src, __bf16* __restrict__ dst,
                         int R, int C) {
    __shared__ float tile[32][33];
    const int e  = blockIdx.z;
    const int r0 = blockIdx.y * 32;
    const int c0 = blockIdx.x * 32;
    const int tx = threadIdx.x;       // 0..31
    const int ty = threadIdx.y;       // 0..7
    const float* sb = src + (size_t)e * R * C;
    __bf16*      db = dst + (size_t)e * C * R;
    #pragma unroll
    for (int i = 0; i < 4; ++i)
        tile[ty + i * 8][tx] = sb[(size_t)(r0 + ty + i * 8) * C + c0 + tx];
    __syncthreads();
    #pragma unroll
    for (int i = 0; i < 4; ++i)
        db[(size_t)(c0 + ty + i * 8) * R + r0 + tx] = (__bf16)tile[tx][ty + i * 8];
}

// GEMM-1 inner: two independent WMMA accumulation chains over 8 preloaded
// B-fragments; returns hacc0+hacc1.
__device__ inline v8f gemm1_chunk(const v16bf* ag, const __bf16* Bbase, int lane) {
    v16bf vb[8];
    #pragma unroll
    for (int kk = 0; kk < 8; ++kk)
        vb[kk] = load_frag_b(Bbase + kk * 32, W1S, lane);

    v8f hacc0, hacc1;
    #pragma unroll
    for (int v = 0; v < 8; ++v) { hacc0[v] = 0.0f; hacc1[v] = 0.0f; }

    #pragma unroll
    for (int kk = 0; kk < 8; kk += 2) {
        hacc0 = __builtin_amdgcn_wmma_f32_16x16x32_bf16(
            false, ag[kk],     false, vb[kk],     (short)0, hacc0, false, false);
        hacc1 = __builtin_amdgcn_wmma_f32_16x16x32_bf16(
            false, ag[kk + 1], false, vb[kk + 1], (short)0, hacc1, false, false);
    }
    #pragma unroll
    for (int v = 0; v < 8; ++v) hacc0[v] += hacc1[v];
    return hacc0;
}

// ---------------------------------------------------------------------------
// Main fused kernel (async path): weights already bf16 + transposed in ws:
//   W1t[e][h][m]  (K=m contiguous)   W2t[e][o][h]  (K=h contiguous)
// LDS staging uses global_load_async_to_lds_b128 (ASYNCcnt).
// ---------------------------------------------------------------------------
__global__ __launch_bounds__(NT)
void rfnet_fused_async(const float* __restrict__ x,
                       const long long* __restrict__ masks,
                       const float* __restrict__ b1,
                       const float* __restrict__ b2,
                       const __bf16* __restrict__ W1t,
                       const __bf16* __restrict__ W2t,
                       float* __restrict__ out) {
    __shared__ __bf16 Gl[BT * GS];      // 32768 B
    __shared__ __bf16 W1l[HC * W1S];    // 16896 B  [n][k]
    __shared__ __bf16 W2l[O_ * W2S];    // 10240 B  [o][k]
    __shared__ __bf16 hl[BT * HS];      //  5120 B  [r][k]

    const int tid  = threadIdx.x;
    const int lane = tid & 31;
    const int wid  = tid >> 5;
    const int wr   = wid & 3;
    const int wc   = wid >> 2;
    const int e    = blockIdx.y;
    const int row0 = blockIdx.x * BT;

    // ---- Gather: Gl[r][m] = (bf16) x[row0+r][ masks[e][m] ] ----
    {
        const int m  = tid;
        const int mi = (int)masks[(size_t)e * M_ + m];
        const float* xp = x + (size_t)row0 * D_ + mi;
        #pragma unroll 4
        for (int r = 0; r < BT; ++r)
            Gl[r * GS + m] = (__bf16)xp[(size_t)r * D_];
    }
    __syncthreads();

    // ---- Cache all GEMM-1 A-fragments in registers (invariant over hc) ----
    v16bf ag[8];
    #pragma unroll
    for (int kk = 0; kk < 8; ++kk)
        ag[kk] = load_frag_a(&Gl[(16 * wr) * GS + kk * 32], GS, lane);

    v8f oacc[4];
    #pragma unroll
    for (int t = 0; t < 4; ++t)
        #pragma unroll
        for (int v = 0; v < 8; ++v) oacc[t][v] = 0.0f;

    const __bf16* W1te = W1t + (size_t)e * H_ * M_;   // [h][m]
    const __bf16* W2te = W2t + (size_t)e * O_ * H_;   // [o][h]
    const unsigned long long w1b = (unsigned long long)(uintptr_t)W1te;
    const unsigned long long w2b = (unsigned long long)(uintptr_t)W2te;
    const unsigned w1l0 = (unsigned)(uintptr_t)&W1l[0];
    const unsigned w2l0 = (unsigned)(uintptr_t)&W2l[0];

    for (int hc = 0; hc < H_; hc += HC) {
        __syncthreads();   // previous chunk's consumers done

        // ---- Async stage W1 chunk: rows n=0..31 of W1t[hc+n][0..255] ----
        // One wave copies one 512B row per instruction (16B per lane).
        #pragma unroll
        for (int j = 0; j < 4; ++j) {
            const int n = wid + j * 8;
            const unsigned lds  = w1l0 + (unsigned)(n * W1S + lane * 8) * 2u;
            const unsigned goff = (unsigned)(((hc + n) * M_ + lane * 8) * 2);
            asm volatile("global_load_async_to_lds_b128 %0, %1, %2"
                         :: "v"(lds), "v"(goff), "s"(w1b) : "memory");
        }
        // ---- Async stage W2 chunk: rows o=0..127 of W2t[o][hc..hc+31] ----
        // 64B per row -> 4 lanes/row, 8 rows per instruction per wave.
        #pragma unroll
        for (int j = 0; j < 2; ++j) {
            const int o = wid * 16 + j * 8 + (lane >> 2);
            const unsigned lds  = w2l0 + (unsigned)(o * W2S + (lane & 3) * 8) * 2u;
            const unsigned goff = (unsigned)((o * H_ + hc + (lane & 3) * 8) * 2);
            asm volatile("global_load_async_to_lds_b128 %0, %1, %2"
                         :: "v"(lds), "v"(goff), "s"(w2b) : "memory");
        }
        if (hc + HC < H_) {  // warm L2 for next chunk (global_prefetch_b8)
            __builtin_prefetch(&W1te[(size_t)(hc + HC) * M_], 0, 1);
            __builtin_prefetch(&W2te[hc + HC], 0, 1);
        }
        asm volatile("s_wait_asynccnt 0x0" ::: "memory");
        __syncthreads();

        // ---- GEMM 1: h(64xHC) = relu( g @ W1chunk + b1 ) ----
        {
            v8f hacc = gemm1_chunk(ag, &W1l[(16 * wc) * W1S], lane);
            const int colL  = 16 * wc + (lane & 15);
            const float bb  = b1[(size_t)e * H_ + hc + colL];
            const int rbase = 16 * wr + ((lane & 16) ? 8 : 0);
            #pragma unroll
            for (int v = 0; v < 8; ++v) {
                float f = fmaxf(hacc[v] + bb, 0.0f);
                hl[(rbase + v) * HS + colL] = (__bf16)f;
            }
        }
        __syncthreads();

        // ---- GEMM 2: oacc += h(64xHC) @ W2chunk(HCx128), K=32 ----
        {
            v16bf va = load_frag_a(&hl[(16 * wr) * HS], HS, lane);
            v16bf vb[4];
            #pragma unroll
            for (int t = 0; t < 4; ++t)
                vb[t] = load_frag_b(&W2l[(16 * (wc * 4 + t)) * W2S], W2S, lane);
            #pragma unroll
            for (int t = 0; t < 4; ++t)
                oacc[t] = __builtin_amdgcn_wmma_f32_16x16x32_bf16(
                    false, va, false, vb[t], (short)0, oacc[t], false, false);
        }
    }

    // ---- Finalize: out += (oacc + b2) / E ----
    const float invE = 1.0f / (float)E_;
    const int rbase  = 16 * wr + ((lane & 16) ? 8 : 0);
    #pragma unroll
    for (int t = 0; t < 4; ++t) {
        const int col  = 16 * (wc * 4 + t) + (lane & 15);
        const float bb = b2[(size_t)e * O_ + col];
        #pragma unroll
        for (int v = 0; v < 8; ++v) {
            float val = (oacc[t][v] + bb) * invE;
            atomicAdd(&out[(size_t)(row0 + rbase + v) * O_ + col], val);
        }
    }
}

// ---------------------------------------------------------------------------
// Fallback (ws too small): converts f32 weights in-kernel.
// ---------------------------------------------------------------------------
__global__ __launch_bounds__(NT)
void rfnet_fused_fallback(const float* __restrict__ x,
                          const long long* __restrict__ masks,
                          const float* __restrict__ W1,
                          const float* __restrict__ b1,
                          const float* __restrict__ W2,
                          const float* __restrict__ b2,
                          float* __restrict__ out) {
    __shared__ __bf16 Gl[BT * GS];
    __shared__ __bf16 W1l[HC * W1S];
    __shared__ __bf16 W2l[O_ * W2S];
    __shared__ __bf16 hl[BT * HS];

    const int tid  = threadIdx.x;
    const int lane = tid & 31;
    const int wid  = tid >> 5;
    const int wr   = wid & 3;
    const int wc   = wid >> 2;
    const int e    = blockIdx.y;
    const int row0 = blockIdx.x * BT;

    {
        const int m  = tid;
        const int mi = (int)masks[(size_t)e * M_ + m];
        const float* xp = x + (size_t)row0 * D_ + mi;
        #pragma unroll 4
        for (int r = 0; r < BT; ++r)
            Gl[r * GS + m] = (__bf16)xp[(size_t)r * D_];
    }
    __syncthreads();

    v16bf ag[8];
    #pragma unroll
    for (int kk = 0; kk < 8; ++kk)
        ag[kk] = load_frag_a(&Gl[(16 * wr) * GS + kk * 32], GS, lane);

    v8f oacc[4];
    #pragma unroll
    for (int t = 0; t < 4; ++t)
        #pragma unroll
        for (int v = 0; v < 8; ++v) oacc[t][v] = 0.0f;

    const float* W1e = W1 + (size_t)e * M_ * H_;
    const float* W2e = W2 + (size_t)e * H_ * O_;

    for (int hc = 0; hc < H_; hc += HC) {
        __syncthreads();
        {
            const int n = tid & (HC - 1);
            #pragma unroll
            for (int i = 0; i < (M_ * HC) / NT; ++i) {
                const int k = (tid >> 5) + i * (NT / HC);
                W1l[n * W1S + k] = (__bf16)W1e[(size_t)k * H_ + hc + n];
            }
        }
        {
            const int o = tid & (O_ - 1);
            #pragma unroll
            for (int i = 0; i < (HC * O_) / NT; ++i) {
                const int k = (tid >> 7) + i * (NT / O_);
                W2l[o * W2S + k] = (__bf16)W2e[(size_t)(hc + k) * O_ + o];
            }
        }
        __syncthreads();
        {
            v8f hacc = gemm1_chunk(ag, &W1l[(16 * wc) * W1S], lane);
            const int colL  = 16 * wc + (lane & 15);
            const float bb  = b1[(size_t)e * H_ + hc + colL];
            const int rbase = 16 * wr + ((lane & 16) ? 8 : 0);
            #pragma unroll
            for (int v = 0; v < 8; ++v) {
                float f = fmaxf(hacc[v] + bb, 0.0f);
                hl[(rbase + v) * HS + colL] = (__bf16)f;
            }
        }
        __syncthreads();
        {
            v16bf va = load_frag_a(&hl[(16 * wr) * HS], HS, lane);
            v16bf vb[4];
            #pragma unroll
            for (int t = 0; t < 4; ++t)
                vb[t] = load_frag_b(&W2l[(16 * (wc * 4 + t)) * W2S], W2S, lane);
            #pragma unroll
            for (int t = 0; t < 4; ++t)
                oacc[t] = __builtin_amdgcn_wmma_f32_16x16x32_bf16(
                    false, va, false, vb[t], (short)0, oacc[t], false, false);
        }
    }

    const float invE = 1.0f / (float)E_;
    const int rbase  = 16 * wr + ((lane & 16) ? 8 : 0);
    #pragma unroll
    for (int t = 0; t < 4; ++t) {
        const int col  = 16 * (wc * 4 + t) + (lane & 15);
        const float bb = b2[(size_t)e * O_ + col];
        #pragma unroll
        for (int v = 0; v < 8; ++v) {
            float val = (oacc[t][v] + bb) * invE;
            atomicAdd(&out[(size_t)(row0 + rbase + v) * O_ + col], val);
        }
    }
}

extern "C" void kernel_launch(void* const* d_in, const int* in_sizes, int n_in,
                              void* d_out, int out_size, void* d_ws, size_t ws_size,
                              hipStream_t stream) {
    (void)in_sizes; (void)n_in;
    const float*     x  = (const float*)d_in[0];
    const long long* mk = (const long long*)d_in[1];
    const float*     W1 = (const float*)d_in[2];
    const float*     b1 = (const float*)d_in[3];
    const float*     W2 = (const float*)d_in[4];
    const float*     b2 = (const float*)d_in[5];
    float* out = (float*)d_out;

    rfnet_zero_out<<<(out_size + NT - 1) / NT, NT, 0, stream>>>(out, out_size);

    const size_t w1t_elems = (size_t)E_ * H_ * M_;
    const size_t w2t_elems = (size_t)E_ * O_ * H_;
    const size_t need = (w1t_elems + w2t_elems) * sizeof(__bf16);

    dim3 grid(B_ / BT, E_);
    if (ws_size >= need) {
        __bf16* W1t = (__bf16*)d_ws;             // [e][h][m]
        __bf16* W2t = W1t + w1t_elems;           // [e][o][h]
        dim3 tb(32, 8);
        rfnet_transpose_cvt<<<dim3(H_ / 32, M_ / 32, E_), tb, 0, stream>>>(W1, W1t, M_, H_);
        rfnet_transpose_cvt<<<dim3(O_ / 32, H_ / 32, E_), tb, 0, stream>>>(W2, W2t, H_, O_);
        rfnet_fused_async<<<grid, NT, 0, stream>>>(x, mk, b1, b2, W1t, W2t, out);
    } else {
        rfnet_fused_fallback<<<grid, NT, 0, stream>>>(x, mk, W1, b1, W2, b2, out);
    }
}
